// GoldenMoELayer_49821620634171
// MI455X (gfx1250) — compile-verified, matrix operand order checked by reading, not run on hardware
//
#include <hip/hip_runtime.h>
#include <hip/hip_bf16.h>

// ---------------------------------------------------------------------------
// GoldenMoE: B=2, T=2048, D=2048, E=8, F=2048  (dense MoE, weighted combine)
// M = B*T = 4096 tokens. 24 GEMMs of 4096x2048x2048 -> bf16 WMMA path.
// CDNA5 paths: v_wmma_f32_16x16x32_bf16, global_load_async_to_lds_b128 (+
// s_wait_asynccnt), global_prefetch.
// ---------------------------------------------------------------------------

constexpr int M  = 4096;   // B*T tokens
constexpr int Dm = 2048;   // model dim
constexpr int Fm = 2048;   // expert intermediate dim
constexpr int E  = 8;      // experts

typedef __attribute__((ext_vector_type(16))) __bf16 v16bf;
typedef __attribute__((ext_vector_type(8)))  __bf16 v8bf;
typedef __attribute__((ext_vector_type(8)))  float  v8f;

// round-to-nearest-even f32 -> bf16 (bit form)
__device__ __forceinline__ unsigned short f2bf_bits(float f) {
    unsigned u = __float_as_uint(f);
    u += 0x7FFFu + ((u >> 16) & 1u);
    return (unsigned short)(u >> 16);
}
__device__ __forceinline__ __bf16 f2bf(float f) {
    unsigned short s = f2bf_bits(f);
    return __builtin_bit_cast(__bf16, s);
}

// LDS byte offset of a shared-memory pointer (HW truncates flat LDS addr)
__device__ __forceinline__ unsigned lds_u32(const void* p) {
    return (unsigned)(reinterpret_cast<uintptr_t>(p));
}

// 64B per-lane async global->LDS copy (4 x b128), tracked by ASYNCcnt
__device__ __forceinline__ void async_copy_64B(unsigned lds_dst, const void* gsrc) {
    asm volatile(
        "global_load_async_to_lds_b128 %0, %1, off\n\t"
        "global_load_async_to_lds_b128 %0, %1, off offset:16\n\t"
        "global_load_async_to_lds_b128 %0, %1, off offset:32\n\t"
        "global_load_async_to_lds_b128 %0, %1, off offset:48"
        :: "v"(lds_dst), "v"(gsrc) : "memory");
}
__device__ __forceinline__ void wait_async0() {
    asm volatile("s_wait_asynccnt 0x0" ::: "memory");
}

// ---------------------------------------------------------------------------
// WMMA tiling parameters
// ---------------------------------------------------------------------------
constexpr int TS   = 64;   // block tile (M and N)
constexpr int KS   = 64;   // K per barrier stage (2 x k32 WMMA chunks)
constexpr int LSTR = 80;   // LDS row stride in halves (160B, 16B-aligned rows)

// A/B fragment load from LDS, 16-bit 16x32 layout:
//  lanes 0-15 : row = row0+lane,    K = {0..7, 16..23}
//  lanes 16-31: row = row0+lane-16, K = {8..15, 24..31}
__device__ __forceinline__ v16bf ld_frag(const __bf16* lds, int row0, int kk, int lane) {
    const int hlf = lane >> 4;
    const int r   = row0 + (lane & 15);
    const __bf16* p = lds + r * LSTR + kk * 32 + hlf * 8;
    v8bf lo = *(const v8bf*)(p);
    v8bf hi = *(const v8bf*)(p + 16);
    v16bf out;
#pragma unroll
    for (int i = 0; i < 8; ++i) { out[i] = lo[i]; out[i + 8] = hi[i]; }
    return out;
}

// Stage a 64(k) x 64(n) fp32 weight tile into LDS as bf16 [n][k] (transposed).
// Thread handles k-pair 2*kp..2*kp+1, n-range seg*16..+15; stores b32 pairs.
__device__ __forceinline__ void stage_b(__bf16* __restrict__ Bs,
                                        const float* __restrict__ W, int ldn,
                                        int kb, int n0, int kp, int seg) {
    const float* r0 = W + (size_t)(kb + 2 * kp) * ldn + n0 + seg * 16;
    const float* r1 = r0 + ldn;
    if (kb + KS < Dm) {  // prefetch next stage's rows (Dm==Fm, same K extent)
        __builtin_prefetch(r0 + (size_t)KS * ldn, 0, 0);
        __builtin_prefetch(r1 + (size_t)KS * ldn, 0, 0);
    }
#pragma unroll
    for (int v = 0; v < 4; ++v) {
        float4 a = ((const float4*)r0)[v];
        float4 b = ((const float4*)r1)[v];
        int nb = seg * 16 + v * 4;
        unsigned p0 = f2bf_bits(a.x) | ((unsigned)f2bf_bits(b.x) << 16);
        unsigned p1 = f2bf_bits(a.y) | ((unsigned)f2bf_bits(b.y) << 16);
        unsigned p2 = f2bf_bits(a.z) | ((unsigned)f2bf_bits(b.z) << 16);
        unsigned p3 = f2bf_bits(a.w) | ((unsigned)f2bf_bits(b.w) << 16);
        *(unsigned*)&Bs[(nb + 0) * LSTR + 2 * kp] = p0;
        *(unsigned*)&Bs[(nb + 1) * LSTR + 2 * kp] = p1;
        *(unsigned*)&Bs[(nb + 2) * LSTR + 2 * kp] = p2;
        *(unsigned*)&Bs[(nb + 3) * LSTR + 2 * kp] = p3;
    }
}

// ---------------------------------------------------------------------------
// x fp32 -> bf16
// ---------------------------------------------------------------------------
__global__ __launch_bounds__(256) void k_cvt_x(const float* __restrict__ x,
                                               __bf16* __restrict__ xb, int n) {
    int i = (blockIdx.x * 256 + threadIdx.x) * 4;
    if (i < n) {
        float4 v = *(const float4*)(x + i);
        xb[i + 0] = f2bf(v.x);
        xb[i + 1] = f2bf(v.y);
        xb[i + 2] = f2bf(v.z);
        xb[i + 3] = f2bf(v.w);
    }
}

__global__ __launch_bounds__(256) void k_zero(float* __restrict__ out, int n) {
    int i = (blockIdx.x * 256 + threadIdx.x) * 4;
    if (i < n) *(float4*)(out + i) = make_float4(0.f, 0.f, 0.f, 0.f);
}

// ---------------------------------------------------------------------------
// Router: golden-zone sigmoid gating with top-2 fallback. One wave per token.
// ---------------------------------------------------------------------------
__global__ __launch_bounds__(256) void k_router(const float* __restrict__ x,
                                                const float* __restrict__ Wr,
                                                const float* __restrict__ temp,
                                                float* __restrict__ wout) {
    const int lane = threadIdx.x & 31;
    const int wv   = threadIdx.x >> 5;
    const int t    = blockIdx.x * 8 + wv;

    float acc[E] = {};
    const float* xr = x + (size_t)t * Dm;
    for (int d = lane; d < Dm; d += 32) {
        float xv = xr[d];
        const float4* w4 = (const float4*)(Wr + (size_t)d * E);
        float4 a = w4[0], b = w4[1];
        acc[0] += xv * a.x; acc[1] += xv * a.y; acc[2] += xv * a.z; acc[3] += xv * a.w;
        acc[4] += xv * b.x; acc[5] += xv * b.y; acc[6] += xv * b.z; acc[7] += xv * b.w;
    }
#pragma unroll
    for (int e = 0; e < E; ++e)
#pragma unroll
        for (int off = 16; off; off >>= 1)
            acc[e] += __shfl_xor(acc[e], off, 32);

    if (lane == 0) {
        const float T0 = temp[0];
        const float CENTER = 0.36787944117144233f;   // 1/e
        const float LOWER  = 0.21231792754821910f;   // 0.5 - ln(4/3)
        float w[E], dist[E], wsum = 0.f;
#pragma unroll
        for (int e = 0; e < E; ++e) {
            float inh = 1.f / (1.f + __expf(-acc[e] / T0));
            float zone = (inh >= LOWER && inh <= 0.5f) ? 1.f : 0.f;
            dist[e] = fabsf(inh - CENTER);
            w[e] = __expf(-dist[e] * 10.f) * zone;
            wsum += w[e];
        }
        if (wsum < 1e-8f) {
            float fb[E];
#pragma unroll
            for (int e = 0; e < E; ++e) fb[e] = __expf(-dist[e] * (1.f / 0.3f));
            int i1 = 0;
            for (int e = 1; e < E; ++e) if (fb[e] > fb[i1]) i1 = e;   // first max
            int i2 = -1;
            for (int e = 0; e < E; ++e) {
                if (e == i1) continue;
                if (i2 < 0 || fb[e] > fb[i2]) i2 = e;                 // second max
            }
            float s = fmaxf(fb[i1] + fb[i2], 1e-8f);
#pragma unroll
            for (int e = 0; e < E; ++e) w[e] = 0.f;
            w[i1] = fb[i1] / s;
            w[i2] = fb[i2] / s;
            wsum  = w[i1] + w[i2];
        }
        wsum = fmaxf(wsum, 1e-8f);
#pragma unroll
        for (int e = 0; e < E; ++e) wout[(size_t)t * E + e] = w[e] / wsum;
    }
}

// ---------------------------------------------------------------------------
// GEMM1 (per expert): h = silu(x @ Wg) * (x @ Wu)      [M x F], bf16 out
//   A = xb [M][D] bf16 (async global->LDS); Bg/Bu = W [D][F] fp32 (cvt in LDS)
// 128 threads = 4 waves; block 64x64, wave 32x32, K=64/stage, 16 WMMA/stage.
// ---------------------------------------------------------------------------
__global__ __launch_bounds__(128) void k_gemm1(const __bf16* __restrict__ xb,
                                               const float* __restrict__ Wg,
                                               const float* __restrict__ Wu,
                                               __bf16* __restrict__ h) {
    __shared__ __bf16 As[TS * LSTR];
    __shared__ __bf16 Bgs[TS * LSTR];
    __shared__ __bf16 Bus[TS * LSTR];

    const int tid  = threadIdx.x;
    const int lane = tid & 31;
    const int wv   = tid >> 5;
    const int n0   = blockIdx.x * TS;
    const int m0   = blockIdx.y * TS;
    const int wm   = (wv >> 1) * 32;
    const int wn   = (wv & 1) * 32;

    v8f accg[2][2] = {};
    v8f accu[2][2] = {};

    const int arow = tid >> 1, aseg = tid & 1;   // A: 64 rows x 2 segs of 32 halves
    const int kp   = tid >> 2, bseg = tid & 3;   // B: 32 k-pairs x 4 segs of 16 n

    for (int kb = 0; kb < Dm; kb += KS) {
        // A tile: 64 rows x 64 halves, async copy straight into LDS
        {
            const __bf16* gsrc = xb + (size_t)(m0 + arow) * Dm + kb + aseg * 32;
            unsigned ldst = lds_u32(As + arow * LSTR + aseg * 32);
            async_copy_64B(ldst, gsrc);
        }
        // B tiles: fp32 -> bf16, transposed [n][k], paired b32 stores
        stage_b(Bgs, Wg, Fm, kb, n0, kp, bseg);
        stage_b(Bus, Wu, Fm, kb, n0, kp, bseg);

        wait_async0();
        __syncthreads();

#pragma unroll
        for (int kk = 0; kk < 2; ++kk) {
            v16bf af[2], bg[2], bu[2];
#pragma unroll
            for (int i = 0; i < 2; ++i) af[i] = ld_frag(As,  wm + i * 16, kk, lane);
#pragma unroll
            for (int j = 0; j < 2; ++j) bg[j] = ld_frag(Bgs, wn + j * 16, kk, lane);
#pragma unroll
            for (int j = 0; j < 2; ++j) bu[j] = ld_frag(Bus, wn + j * 16, kk, lane);
#pragma unroll
            for (int i = 0; i < 2; ++i)
#pragma unroll
                for (int j = 0; j < 2; ++j) {
                    accg[i][j] = __builtin_amdgcn_wmma_f32_16x16x32_bf16(
                        false, af[i], false, bg[j], (short)0, accg[i][j], false, false);
                    accu[i][j] = __builtin_amdgcn_wmma_f32_16x16x32_bf16(
                        false, af[i], false, bu[j], (short)0, accu[i][j], false, false);
                }
        }
        __syncthreads();
    }

    // epilogue: h = silu(g) * u  (C/D layout: lane<16 -> M=r, lane>=16 -> M=r+8)
    const int hlf = lane >> 4, ln = lane & 15;
#pragma unroll
    for (int i = 0; i < 2; ++i)
#pragma unroll
        for (int j = 0; j < 2; ++j)
#pragma unroll
            for (int r = 0; r < 8; ++r) {
                int row = m0 + wm + i * 16 + r + hlf * 8;
                int col = n0 + wn + j * 16 + ln;
                float g = accg[i][j][r];
                float u = accu[i][j][r];
                float s = g * __builtin_amdgcn_rcpf(1.f + __expf(-g));
                h[(size_t)row * Fm + col] = f2bf(s * u);
            }
}

// ---------------------------------------------------------------------------
// GEMM2 (per expert e): out[t][d] += w[t][e] * (h @ Wd)[t][d]
// ---------------------------------------------------------------------------
__global__ __launch_bounds__(128) void k_gemm2(const __bf16* __restrict__ h,
                                               const float* __restrict__ Wd,
                                               const float* __restrict__ wr,
                                               int e,
                                               float* __restrict__ out) {
    __shared__ __bf16 As[TS * LSTR];
    __shared__ __bf16 Bs[TS * LSTR];

    const int tid  = threadIdx.x;
    const int lane = tid & 31;
    const int wv   = tid >> 5;
    const int n0   = blockIdx.x * TS;
    const int m0   = blockIdx.y * TS;
    const int wm   = (wv >> 1) * 32;
    const int wn   = (wv & 1) * 32;

    v8f acc[2][2] = {};

    const int arow = tid >> 1, aseg = tid & 1;
    const int kp   = tid >> 2, bseg = tid & 3;

    for (int kb = 0; kb < Fm; kb += KS) {
        {
            const __bf16* gsrc = h + (size_t)(m0 + arow) * Fm + kb + aseg * 32;
            unsigned ldst = lds_u32(As + arow * LSTR + aseg * 32);
            async_copy_64B(ldst, gsrc);
        }
        stage_b(Bs, Wd, Dm, kb, n0, kp, bseg);

        wait_async0();
        __syncthreads();

#pragma unroll
        for (int kk = 0; kk < 2; ++kk) {
            v16bf af[2], bf[2];
#pragma unroll
            for (int i = 0; i < 2; ++i) af[i] = ld_frag(As, wm + i * 16, kk, lane);
#pragma unroll
            for (int j = 0; j < 2; ++j) bf[j] = ld_frag(Bs, wn + j * 16, kk, lane);
#pragma unroll
            for (int i = 0; i < 2; ++i)
#pragma unroll
                for (int j = 0; j < 2; ++j)
                    acc[i][j] = __builtin_amdgcn_wmma_f32_16x16x32_bf16(
                        false, af[i], false, bf[j], (short)0, acc[i][j], false, false);
        }
        __syncthreads();
    }

    const int hlf = lane >> 4, ln = lane & 15;
#pragma unroll
    for (int i = 0; i < 2; ++i)
#pragma unroll
        for (int j = 0; j < 2; ++j)
#pragma unroll
            for (int r = 0; r < 8; ++r) {
                int row = m0 + wm + i * 16 + r + hlf * 8;
                int col = n0 + wn + j * 16 + ln;
                float wgt = wr[(size_t)row * E + e];
                out[(size_t)row * Dm + col] += wgt * acc[i][j][r];
            }
}

// ---------------------------------------------------------------------------
// Launcher: cvt -> router -> zero -> per-expert {gemm1, gemm2} (serialized)
// Workspace: xb(16.8MB) | h(16.8MB) | router weights(128KB)
// ---------------------------------------------------------------------------
extern "C" void kernel_launch(void* const* d_in, const int* in_sizes, int n_in,
                              void* d_out, int out_size, void* d_ws, size_t ws_size,
                              hipStream_t stream) {
    const float* x    = (const float*)d_in[0];
    const float* Wg   = (const float*)d_in[1];
    const float* Wu   = (const float*)d_in[2];
    const float* Wd   = (const float*)d_in[3];
    const float* Wr   = (const float*)d_in[4];
    const float* temp = (const float*)d_in[5];
    float* out = (float*)d_out;

    char* ws = (char*)d_ws;
    __bf16* xb = (__bf16*)ws;                                  // M*D bf16
    __bf16* hb = (__bf16*)(ws + (size_t)M * Dm * 2);           // M*F bf16
    float*  wr = (float*)(ws + (size_t)M * Dm * 2 * 2);        // M*E fp32

    const int nx = M * Dm;
    k_cvt_x<<<nx / 4 / 256, 256, 0, stream>>>(x, xb, nx);
    k_router<<<M / 8, 256, 0, stream>>>(x, Wr, temp, wr);
    k_zero<<<out_size / 4 / 256 + 1, 256, 0, stream>>>(out, out_size);

    dim3 blk(128);
    dim3 g1(Fm / TS, M / TS);
    dim3 g2(Dm / TS, M / TS);
    for (int e = 0; e < E; ++e) {
        k_gemm1<<<g1, blk, 0, stream>>>(xb, Wg + (size_t)e * Dm * Fm,
                                        Wu + (size_t)e * Dm * Fm, hb);
        k_gemm2<<<g2, blk, 0, stream>>>(hb, Wd + (size_t)e * Fm * Dm, wr, e, out);
    }
}
